// GeoMapNet_72043781423753
// MI455X (gfx1250) — compile-verified, matrix operand order checked by reading, not run on hardware
//
#include <hip/hip_runtime.h>
#include <math.h>

// ---------------------------------------------------------------------------
// Fused 8-layer transformer encoder for MI455X (gfx1250, wave32, WMMA).
// All GEMMs run as v_wmma_f32_16x16x32_f16 (f16 in, f32 accumulate).
// Weights are pre-converted to f16, zero-padded to N%16==0 / K%32==0 in d_ws,
// with biases folded into a spare K column (activations carry a 1.0 there).
// One workgroup = 32 batch elements (96 rows), 4 waves, activations in LDS.
// GEMM jobs are N-tile *pairs*: one A fragment feeds two independent
// v_wmma accumulator chains (ILP for the 1-wave-per-SIMD regime).
// ---------------------------------------------------------------------------

#define LAYERS 8
#define DM     100
#define FF     512
#define NW     4        // waves per block
#define NT_THR 128      // threads per block

// f16-element offsets inside the workspace weight buffer
#define OFF_WE   0            // We padded [112 x 96], bias/pe in cols 72..75
#define OFF_WF   10752        // Wf padded [48 x 320], bias in col 300
#define OFF_L0   26112
#define L_STRIDE 180224       // per-layer block
#define LOFF_WQ  0            // [112 x 128], bias in col 100
#define LOFF_WK  14336
#define LOFF_WV  28672
#define LOFF_WO  43008        // [112 x 128], no folded bias
#define LOFF_W1  57344        // [512 x 128], bias in col 100
#define LOFF_W2  122880       // [112 x 512], no folded bias

#define Z8 {0.f,0.f,0.f,0.f,0.f,0.f,0.f,0.f}

typedef __attribute__((ext_vector_type(16))) _Float16     v16h;
typedef __attribute__((ext_vector_type(8)))  float        v8f;
typedef __attribute__((ext_vector_type(4)))  unsigned int v4u;

union Frag { v16h v; v4u q[2]; };

// ---------------------------------------------------------------------------
// fp32 -> f16 weight conversion, zero padded; optional bias folded at col K
// ---------------------------------------------------------------------------
__global__ void convert_pad_kernel(const float* __restrict__ src,
                                   const float* __restrict__ bias,
                                   _Float16* __restrict__ dst,
                                   int N, int K, int Np, int Kp) {
  int idx = blockIdx.x * blockDim.x + threadIdx.x;
  if (idx >= Np * Kp) return;
  int n = idx / Kp, k = idx % Kp;
  float v = 0.0f;
  if (n < N) {
    if (k < K)                 v = src[n * K + k];
    else if (k == K && bias)   v = bias[n];
  }
  dst[idx] = (_Float16)v;
}

// Embedding weight [112 x 96]: cols 0..71 = We, col 72 = be, cols 73..75 = pe
__global__ void build_embed_kernel(const float* __restrict__ We,
                                   const float* __restrict__ be,
                                   const float* __restrict__ pe,
                                   _Float16* __restrict__ dst) {
  int idx = blockIdx.x * blockDim.x + threadIdx.x;
  if (idx >= 112 * 96) return;
  int n = idx / 96, k = idx % 96;
  float v = 0.0f;
  if (n < DM) {
    if (k < 72)                  v = We[n * 72 + k];
    else if (k == 72)            v = be[n];
    else if (k <= 75)            v = pe[(k - 73) * DM + n];
  }
  dst[idx] = (_Float16)v;
}

// ---------------------------------------------------------------------------
// WMMA tile helpers (fragment layouts per ISA 7.12.2)
// ---------------------------------------------------------------------------
template <int KS>
__device__ __forceinline__ v8f tile_mm(const _Float16* A, int as, int mbase,
                                       const _Float16* W, int kpad, int nbase,
                                       int lane, v8f acc) {
  const int l15  = lane & 15;
  const int koff = (lane & 16) ? 8 : 0;
  const _Float16* pa = A + (mbase + l15) * as + koff;
  const _Float16* pw = W + (nbase + l15) * kpad + koff;
#pragma unroll
  for (int ks = 0; ks < KS; ++ks) {
    Frag a, b;
    a.q[0] = *(const v4u*)(pa);
    a.q[1] = *(const v4u*)(pa + 16);
    b.q[0] = *(const v4u*)(pw);
    b.q[1] = *(const v4u*)(pw + 16);
    acc = __builtin_amdgcn_wmma_f32_16x16x32_f16(
        false, a.v, false, b.v, (short)0, acc, false, false);
    pa += 32; pw += 32;
  }
  return acc;
}

// One A fragment feeding two adjacent N-tiles: two independent wmma chains
template <int KS>
__device__ __forceinline__ void tile_mm2(const _Float16* A, int as, int mbase,
                                         const _Float16* W, int kpad, int nbase0,
                                         int lane, v8f& acc0, v8f& acc1) {
  const int l15  = lane & 15;
  const int koff = (lane & 16) ? 8 : 0;
  const _Float16* pa  = A + (mbase + l15) * as + koff;
  const _Float16* pw0 = W + (nbase0 + l15) * kpad + koff;
  const _Float16* pw1 = pw0 + 16 * kpad;
#pragma unroll
  for (int ks = 0; ks < KS; ++ks) {
    Frag a, b0, b1;
    a.q[0]  = *(const v4u*)(pa);
    a.q[1]  = *(const v4u*)(pa + 16);
    b0.q[0] = *(const v4u*)(pw0);
    b0.q[1] = *(const v4u*)(pw0 + 16);
    b1.q[0] = *(const v4u*)(pw1);
    b1.q[1] = *(const v4u*)(pw1 + 16);
    acc0 = __builtin_amdgcn_wmma_f32_16x16x32_f16(
        false, a.v, false, b0.v, (short)0, acc0, false, false);
    acc1 = __builtin_amdgcn_wmma_f32_16x16x32_f16(
        false, a.v, false, b1.v, (short)0, acc1, false, false);
    pa += 32; pw0 += 32; pw1 += 32;
  }
}

// D-tile layout: lane holds N = lane&15, rows mbase + r + 8*(lane>=16)
__device__ __forceinline__ void store_tile_f16(_Float16* dst, int ds,
    int mbase, int nbase, int lane, v8f acc, bool relu, int one_col) {
  int n  = nbase + (lane & 15);
  int mo = mbase + ((lane & 16) ? 8 : 0);
#pragma unroll
  for (int r = 0; r < 8; ++r) {
    float v = acc[r];
    if (relu) v = fmaxf(v, 0.0f);
    if (n == one_col) v = 1.0f;
    dst[(mo + r) * ds + n] = (_Float16)v;
  }
}

__device__ __forceinline__ void store_tile_resid(float* dst, int ds,
    const float* __restrict__ bias, const _Float16* resid, int rs,
    int mbase, int nbase, int lane, v8f acc) {
  int n = nbase + (lane & 15);
  if (n < DM) {
    float bv = bias[n];
    int mo = mbase + ((lane & 16) ? 8 : 0);
#pragma unroll
    for (int r = 0; r < 8; ++r) {
      int m = mo + r;
      dst[m * ds + n] = acc[r] + bv + (float)resid[m * rs + n];
    }
  }
}

// GEMM -> f16 LDS destination (bias pre-folded into weights).
template <int KS>
__device__ __forceinline__ void gemm_f16(const _Float16* A, int as,
    const _Float16* W, int kpad, int ntiles, bool relu, int one_col,
    _Float16* dst, int ds, int wave, int lane) {
  const int npairs = ntiles >> 1;
  for (int p = wave; p < 6 * npairs; p += NW) {
    int mt = p / npairs, nt0 = (p - mt * npairs) * 2;
    v8f a0 = Z8, a1 = Z8;
    tile_mm2<KS>(A, as, mt * 16, W, kpad, nt0 * 16, lane, a0, a1);
    store_tile_f16(dst, ds, mt * 16, nt0 * 16, lane, a0, relu, one_col);
    store_tile_f16(dst, ds, mt * 16, (nt0 + 1) * 16, lane, a1, relu, one_col);
  }
  for (int mt = (ntiles & 1) ? wave : 6; mt < 6; mt += NW) {
    v8f acc = Z8;
    acc = tile_mm<KS>(A, as, mt * 16, W, kpad, (ntiles - 1) * 16, lane, acc);
    store_tile_f16(dst, ds, mt * 16, (ntiles - 1) * 16, lane, acc, relu, one_col);
  }
}

// GEMM + bias + residual -> f32 LDS scratch (pre-LayerNorm values)
template <int KS>
__device__ __forceinline__ void gemm_resid_f32(const _Float16* A, int as,
    const _Float16* W, int kpad, int ntiles,
    const float* __restrict__ bias, const _Float16* resid, int rs,
    float* dst, int ds, int wave, int lane) {
  const int npairs = ntiles >> 1;
  for (int p = wave; p < 6 * npairs; p += NW) {
    int mt = p / npairs, nt0 = (p - mt * npairs) * 2;
    v8f a0 = Z8, a1 = Z8;
    tile_mm2<KS>(A, as, mt * 16, W, kpad, nt0 * 16, lane, a0, a1);
    store_tile_resid(dst, ds, bias, resid, rs, mt * 16, nt0 * 16, lane, a0);
    store_tile_resid(dst, ds, bias, resid, rs, mt * 16, (nt0 + 1) * 16, lane, a1);
  }
  for (int mt = (ntiles & 1) ? wave : 6; mt < 6; mt += NW) {
    v8f acc = Z8;
    acc = tile_mm<KS>(A, as, mt * 16, W, kpad, (ntiles - 1) * 16, lane, acc);
    store_tile_resid(dst, ds, bias, resid, rs, mt * 16, (ntiles - 1) * 16, lane, acc);
  }
}

// LayerNorm rows; also plants the 1.0 in col 100 used for bias folding
__device__ __forceinline__ void layer_norm(const float* Sr, int ss,
    const float* __restrict__ g, const float* __restrict__ b,
    _Float16* dst, int ds, int tid) {
  for (int row = tid; row < 96; row += NT_THR) {
    const float* p = Sr + row * ss;
    float m = 0.f;
    for (int n = 0; n < DM; ++n) m += p[n];
    m *= (1.0f / DM);
    float var = 0.f;
    for (int n = 0; n < DM; ++n) { float d = p[n] - m; var += d * d; }
    var *= (1.0f / DM);
    float inv = rsqrtf(var + 1e-5f);
    _Float16* q = dst + row * ds;
    for (int n = 0; n < DM; ++n)
      q[n] = (_Float16)((p[n] - m) * inv * g[n] + b[n]);
    q[DM] = (_Float16)1.0f;   // bias-folding column
  }
}

// ---------------------------------------------------------------------------
// Fused encoder: grid = 2048 blocks x 128 threads (4 waves)
// ---------------------------------------------------------------------------
__global__ __launch_bounds__(NT_THR) void encoder_fused(
    const float* __restrict__ x,
    const float* __restrict__ bo, const float* __restrict__ ln1g,
    const float* __restrict__ ln1b, const float* __restrict__ b2,
    const float* __restrict__ ln2g, const float* __restrict__ ln2b,
    const _Float16* __restrict__ wbuf, float* __restrict__ out) {

  __shared__ __align__(16) unsigned char smem[195072];
  _Float16* Xs = (_Float16*)(smem);              // [96][136] MID (post-LN1)
  _Float16* Ys = (_Float16*)(smem + 26112);      // [96][136] IN  (layer input)
  unsigned char* Rr = smem + 52224;              // 99840-byte reused region
  _Float16* Qs = (_Float16*)(Rr);                // [96][112]
  _Float16* Ks = (_Float16*)(Rr + 21504);        // [96][112]
  _Float16* Vs = (_Float16*)(Rr + 43008);        // [96][112]
  _Float16* Cs = (_Float16*)(Rr + 64512);        // [96][136] attention ctx
  _Float16* Hs = (_Float16*)(Rr);                // [96][520] FFN hidden
  _Float16* Fs = (_Float16*)(Rr);                // [32][328] classifier stage
  float*    Ss = (float*)(smem + 152064);        // [96][112] f32 pre-LN

  const int tid  = threadIdx.x;
  const int wave = tid >> 5, lane = tid & 31;
  const int b0   = blockIdx.x * 32;

  // zero LDS once (keeps padding columns finite: 0 * zero-weight stays 0)
  for (int i = tid; i < 195072 / 4; i += NT_THR) ((unsigned int*)smem)[i] = 0u;
  __syncthreads();

  // stage input rows (96 rows x 72 cols, fp32 -> f16) into Xs,
  // plus the bias-folding one-hots: col 72 = 1, col 73+s = 1
  for (int i = tid; i < 96 * 72; i += NT_THR) {
    int m = i / 72, c = i % 72;
    Xs[m * 136 + c] = (_Float16)x[(size_t)(b0 * 3 + m) * 72 + c];
  }
  for (int m = tid; m < 96; m += NT_THR) {
    Xs[m * 136 + 72]           = (_Float16)1.0f;
    Xs[m * 136 + 73 + (m % 3)] = (_Float16)1.0f;
  }
  __syncthreads();

  // ---- embedding: Ys = Xs @ We^T (+be+pe folded); col 100 set to 1.0 ----
  gemm_f16<3>(Xs, 136, wbuf + OFF_WE, 96, 7, false, DM, Ys, 136, wave, lane);
  __syncthreads();

  const float scale = 0.31622776601683794f;  // DH^-0.5

  for (int li = 0; li < LAYERS; ++li) {
    const _Float16* wl = wbuf + OFF_L0 + (size_t)li * L_STRIDE;

    // ---- Q, K, V  (input = Ys; biases folded via col 100) ----
    gemm_f16<4>(Ys, 136, wl + LOFF_WQ, 128, 7, false, -1, Qs, 112, wave, lane);
    gemm_f16<4>(Ys, 136, wl + LOFF_WK, 128, 7, false, -1, Ks, 112, wave, lane);
    gemm_f16<4>(Ys, 136, wl + LOFF_WV, 128, 7, false, -1, Vs, 112, wave, lane);
    __syncthreads();

    // re-zero C padding cols 100..127 (H overlapped this region last layer)
    for (int i = tid; i < 96 * 28; i += NT_THR) {
      int r = i / 28, c = 100 + i % 28;
      Cs[r * 136 + c] = (_Float16)0.0f;
    }

    // ---- attention, faithful to the reference's interleaving reshape:
    // head h, query q, dim d <-> per-batch flat offset h*30 + q*10 + d ----
    for (int t = tid; t < 320; t += NT_THR) {
      int lb = t / 10, h = t % 10;
      int rb = lb * 3;
      float qv[3][10], kv[3][10], vv[3][10];
      int rr[3], cc[3];
      for (int s = 0; s < 3; ++s) {
        int off = h * 30 + s * 10;
        int r = rb + off / 100, c = off % 100;
        rr[s] = r; cc[s] = c;
        for (int d = 0; d < 10; ++d) {
          qv[s][d] = (float)Qs[r * 112 + c + d];
          kv[s][d] = (float)Ks[r * 112 + c + d];
          vv[s][d] = (float)Vs[r * 112 + c + d];
        }
      }
      for (int qi = 0; qi < 3; ++qi) {
        float sc[3];
        for (int ki = 0; ki < 3; ++ki) {
          float dt = 0.f;
          for (int d = 0; d < 10; ++d) dt += qv[qi][d] * kv[ki][d];
          sc[ki] = dt * scale;
        }
        float mx = fmaxf(sc[0], fmaxf(sc[1], sc[2]));
        float e0 = expf(sc[0] - mx), e1 = expf(sc[1] - mx), e2 = expf(sc[2] - mx);
        float inv = 1.0f / (e0 + e1 + e2);
        e0 *= inv; e1 *= inv; e2 *= inv;
        for (int d = 0; d < 10; ++d) {
          float cx = e0 * vv[0][d] + e1 * vv[1][d] + e2 * vv[2][d];
          Cs[rr[qi] * 136 + cc[qi] + d] = (_Float16)cx;
        }
      }
    }
    __syncthreads();

    // ---- Wo + bias + residual(Ys) -> f32 scratch, then LN1 -> Xs ----
    gemm_resid_f32<4>(Cs, 136, wl + LOFF_WO, 128, 7, bo + li * DM, Ys, 136, Ss, 112, wave, lane);
    __syncthreads();
    layer_norm(Ss, 112, ln1g + li * DM, ln1b + li * DM, Xs, 136, tid);
    __syncthreads();

    // ---- FFN1: H = relu(Xs @ W1^T) with b1 folded via col 100 ----
    gemm_f16<4>(Xs, 136, wl + LOFF_W1, 128, 32, true, -1, Hs, 520, wave, lane);
    __syncthreads();

    // ---- FFN2 + bias + residual(Xs) -> f32 scratch, then LN2 -> Ys ----
    gemm_resid_f32<16>(Hs, 520, wl + LOFF_W2, 512, 7, b2 + li * DM, Xs, 136, Ss, 112, wave, lane);
    __syncthreads();
    layer_norm(Ss, 112, ln2g + li * DM, ln2b + li * DM, Ys, 136, tid);
    __syncthreads();
  }

  // ---- classifier: stage 300 features per batch + 1.0 at col 300 ----
  for (int i = tid; i < 32 * 328; i += NT_THR) {
    int lb = i / 328, c = i % 328;
    _Float16 v = (_Float16)0.0f;
    if (c < 300)       v = Ys[(lb * 3 + c / 100) * 136 + (c % 100)];
    else if (c == 300) v = (_Float16)1.0f;     // bf folded into Wf col 300
    Fs[lb * 328 + c] = v;
  }
  __syncthreads();
  {
    const _Float16* Wfp = wbuf + OFF_WF;
    for (int j = wave; j < 2 * 3; j += NW) {
      int mt = j / 3, nt = j - mt * 3;
      int mbase = mt * 16;
      v8f acc = Z8;
      acc = tile_mm<10>(Fs, 328, mbase, Wfp, 320, nt * 16, lane, acc);
      int n = nt * 16 + (lane & 15);
      if (n < 40) {
        int mo = mbase + ((lane & 16) ? 8 : 0);
#pragma unroll
        for (int r = 0; r < 8; ++r) {
          int m = mo + r;
          out[(size_t)(b0 + m) * 40 + n] = acc[r];
        }
      }
    }
  }
}

// ---------------------------------------------------------------------------
extern "C" void kernel_launch(void* const* d_in, const int* in_sizes, int n_in,
                              void* d_out, int out_size, void* d_ws, size_t ws_size,
                              hipStream_t stream) {
  (void)in_sizes; (void)n_in; (void)out_size; (void)ws_size;
  const float* x    = (const float*)d_in[0];
  const float* We   = (const float*)d_in[3];
  const float* be   = (const float*)d_in[4];
  const float* pe   = (const float*)d_in[5];
  const float* Wq   = (const float*)d_in[6];
  const float* bq   = (const float*)d_in[7];
  const float* Wk   = (const float*)d_in[8];
  const float* bk   = (const float*)d_in[9];
  const float* Wv   = (const float*)d_in[10];
  const float* bv   = (const float*)d_in[11];
  const float* Wo   = (const float*)d_in[12];
  const float* bo   = (const float*)d_in[13];
  const float* ln1g = (const float*)d_in[14];
  const float* ln1b = (const float*)d_in[15];
  const float* W1   = (const float*)d_in[16];
  const float* b1   = (const float*)d_in[17];
  const float* W2   = (const float*)d_in[18];
  const float* b2   = (const float*)d_in[19];
  const float* ln2g = (const float*)d_in[20];
  const float* ln2b = (const float*)d_in[21];
  const float* Wf   = (const float*)d_in[22];
  const float* bf   = (const float*)d_in[23];
  _Float16* wbuf = (_Float16*)d_ws;   // needs ~2.94 MB of workspace
  float* out = (float*)d_out;

  auto conv = [&](const float* src, const float* bias, size_t off,
                  int N, int K, int Np, int Kp) {
    int total = Np * Kp;
    convert_pad_kernel<<<(total + 255) / 256, 256, 0, stream>>>(
        src, bias, wbuf + off, N, K, Np, Kp);
  };
  build_embed_kernel<<<(112 * 96 + 255) / 256, 256, 0, stream>>>(
      We, be, pe, wbuf + OFF_WE);
  conv(Wf, bf, OFF_WF, 40, 300, 48, 320);
  for (int i = 0; i < LAYERS; ++i) {
    size_t lb = OFF_L0 + (size_t)i * L_STRIDE;
    conv(Wq + (size_t)i * DM * DM, bq + (size_t)i * DM, lb + LOFF_WQ, 100, 100, 112, 128);
    conv(Wk + (size_t)i * DM * DM, bk + (size_t)i * DM, lb + LOFF_WK, 100, 100, 112, 128);
    conv(Wv + (size_t)i * DM * DM, bv + (size_t)i * DM, lb + LOFF_WV, 100, 100, 112, 128);
    conv(Wo + (size_t)i * DM * DM, (const float*)nullptr, lb + LOFF_WO, 100, 100, 112, 128);
    conv(W1 + (size_t)i * FF * DM, b1 + (size_t)i * FF, lb + LOFF_W1, 512, 100, 512, 128);
    conv(W2 + (size_t)i * DM * FF, (const float*)nullptr, lb + LOFF_W2, 100, 512, 112, 512);
  }

  encoder_fused<<<2048, NT_THR, 0, stream>>>(
      x, bo, ln1g, ln1b, b2, ln2g, ln2b, wbuf, out);
}
